// BoT_MHSA_16131897164366
// MI455X (gfx1250) — matrix operand order repaired and checked by reading, not compiled, not run on hardware
//
#include <hip/hip_runtime.h>
#include <hip/hip_bf16.h>

// ---------------------------------------------------------------------------
// BoT MHSA for MI455X (gfx1250, wave32, WMMA bf16).
// B=4, C=256, H=W=64 -> nh=4, hs=64, S=4096.
//
//   0) prep: weights -> bf16 ; x -> bf16 transposed xT[b][p][c]
//   1) proj_kernel: pure-WMMA GEMM. Q scaled by log2(e) (softmax in 2^x
//      domain). Q,K -> [head][s][d] bf16 (K gets rel-pos folded in),
//      V -> [head][d][s] bf16 (pre-transposed).
//   2) attn_kernel: flash attention, transposed formulation, 64-key tiles,
//      double-buffered GLOBAL_LOAD_ASYNC_TO_LDS_B128 pipeline (ASYNCcnt):
//      S^T = K * Q^T   (M=key, N=query)  -> softmax stats live at lane=q
//      O^T = V^T * P   (M=d,   N=query)
// ---------------------------------------------------------------------------

typedef __attribute__((ext_vector_type(16))) __bf16 v16bf;
typedef __attribute__((ext_vector_type(8)))  float  v8f;
typedef __attribute__((ext_vector_type(4)))  int    v4i;

#if defined(__has_builtin)
#  if __has_builtin(__builtin_amdgcn_global_load_async_to_lds_b128)
#    define USE_ASYNC_LDS 1
#  endif
#endif
#ifndef USE_ASYNC_LDS
#  define USE_ASYNC_LDS 0
#endif

__device__ __forceinline__ unsigned short f2bf(float f) {
    return __builtin_bit_cast(unsigned short, (__bf16)f);   // hw cvt, RNE
}

union BF16x16 {
    unsigned short s[16];
    uint4 q[2];
    v16bf v;
};

#if USE_ASYNC_LDS
typedef __attribute__((address_space(1))) v4i* gptr4;   // global int4*
typedef __attribute__((address_space(3))) v4i* lptr4;   // LDS int4* (32-bit)

__device__ __forceinline__ void async_cp16(const unsigned short* g,
                                           unsigned short* l) {
    // flat->LDS: low 32 bits of a flat LDS address ARE the LDS address
    __builtin_amdgcn_global_load_async_to_lds_b128(
        (gptr4)(unsigned long long)g,
        (lptr4)(unsigned int)(unsigned long long)l,
        0, 0);
}
__device__ __forceinline__ void async_wait0() {
#if __has_builtin(__builtin_amdgcn_s_wait_asynccnt)
    __builtin_amdgcn_s_wait_asynccnt(0);
#else
    asm volatile("s_wait_asynccnt 0" ::: "memory");
#endif
}
#endif

// ---------------------------------------------------------------------------
// Prep: fp32 -> bf16 (weights)
// ---------------------------------------------------------------------------
__global__ __launch_bounds__(256)
void cvt_kernel(const float* __restrict__ src, unsigned short* __restrict__ dst,
                int n) {
    const int i = blockIdx.x * 256 + threadIdx.x;
    if (i < n) dst[i] = f2bf(src[i]);
}

// Prep: x[b][c][p] fp32 -> xT[b][p][c] bf16 (output-coalesced)
__global__ __launch_bounds__(256)
void xpose_kernel(const float* __restrict__ x, unsigned short* __restrict__ xT) {
    const long idx = (long)blockIdx.x * 256 + threadIdx.x;  // over [b][p][c]
    const int  c   = (int)(idx & 255);
    const long bp  = idx >> 8;
    const int  p   = (int)(bp & 4095);
    const int  b   = (int)(bp >> 12);
    xT[idx] = f2bf(x[((long)b * 256 + c) * 4096 + p]);
}

// ---------------------------------------------------------------------------
// Projection: Y[o,p] = sum_c W[o,c] * X[c,p] + bias[o]   (per batch)
// grid = (16 o-tiles, 32 p-groups of 128), block = 256 (8 waves).
// mode: 0 = Q (*log2e) -> [n][s][d] ; 1 = K + rel-pos -> [n][s][d]
//       2 = V -> [n][d][s]
// Mapping: f = o*4096 + p ; s = o*16 + (p>>8) ; n = (p>>6)&3 ; d = p&63.
// ---------------------------------------------------------------------------
__global__ __launch_bounds__(256)
void proj_kernel(const unsigned short* __restrict__ xT,   // [4096][256] bf16
                 const unsigned short* __restrict__ wb,   // [256][256]  bf16
                 const float* __restrict__ bias,          // [256]
                 const float* __restrict__ rel_h,         // [4][64][64]
                 const float* __restrict__ rel_w,         // [4][64][64]
                 int mode,
                 unsigned short* __restrict__ out)        // one batch
{
    const int lane = threadIdx.x & 31;
    const int wv   = threadIdx.x >> 5;
    const int hi   = (lane >> 4) & 1;
    const int ln   = lane & 15;

    const int otile = blockIdx.x;                       // 0..15
    const int p     = blockIdx.y * 128 + wv * 16 + ln;  // pixel 0..4095

    v8f acc = {};
    #pragma unroll 2
    for (int kc = 0; kc < 256; kc += 32) {
        // A: weights, M=o row = ln, k(e) = hi*16 + e (contiguous c)
        BF16x16 A;
        const unsigned short* wr = wb + (otile * 16 + ln) * 256 + kc + hi * 16;
        A.q[0] = *(const uint4*)(wr);
        A.q[1] = *(const uint4*)(wr + 8);
        // B: xT row, N=p col = ln, k(e) = hi*16 + e (contiguous c)
        BF16x16 Bv;
        const unsigned short* xr = xT + (long)p * 256 + kc + hi * 16;
        Bv.q[0] = *(const uint4*)(xr);
        Bv.q[1] = *(const uint4*)(xr + 8);

        acc = __builtin_amdgcn_wmma_f32_16x16x32_bf16(
            false, A.v, false, Bv.v, (short)0, acc, false, false);
    }

    // Epilogue: C layout lane=(p col), VGPR r -> row o = otile*16 + r + 8*hi
    const int nh_ = (p >> 6) & 3;
    const int d   = p & 63;
    #pragma unroll
    for (int r = 0; r < 8; ++r) {
        const int o = otile * 16 + r + 8 * hi;
        float val = acc[r] + bias[o];
        const int s = o * 16 + (p >> 8);          // sequence index
        if (mode == 0) val *= 1.4426950408889634f;   // fold log2(e) into Q
        if (mode == 1) {
            val += rel_h[nh_ * 4096 + d * 64 + (s & 63)]
                 + rel_w[nh_ * 4096 + d * 64 + (s >> 6)];
        }
        if (mode == 2) {
            out[(nh_ * 64 + d) * 4096 + s] = f2bf(val);   // V transposed
        } else {
            out[(nh_ * 4096 + s) * 64 + d] = f2bf(val);
        }
    }
}

// ---------------------------------------------------------------------------
// Stage one 64-key tile: K [64][64] contiguous rows, V [d][64] strided rows.
// 8 x 16B chunks per thread (128 threads cover 2 x 8 KB).
// ---------------------------------------------------------------------------
__device__ __forceinline__ void stage_tiles(const unsigned short* __restrict__ Kh,
                                            const unsigned short* __restrict__ Vh,
                                            int kb, int tid,
                                            unsigned short* Kbuf,
                                            unsigned short* Vbuf) {
    const unsigned short* srcK = Kh + (long)kb * 64;
#if USE_ASYNC_LDS
    #pragma unroll
    for (int i = 0; i < 4; ++i) {
        const int c = tid + i * 128;                 // 512 chunks of 16B
        async_cp16(srcK + c * 8, Kbuf + c * 8);
    }
    #pragma unroll
    for (int i = 0; i < 4; ++i) {
        const int u   = tid + i * 128;
        const int d   = u >> 3;                      // 0..63
        const int seg = u & 7;                       // 8-half segment
        async_cp16(Vh + (long)d * 4096 + kb + seg * 8, Vbuf + d * 64 + seg * 8);
    }
#else
    #pragma unroll
    for (int i = 0; i < 4; ++i) {
        const int c = tid + i * 128;
        *(uint4*)(Kbuf + c * 8) = *(const uint4*)(srcK + c * 8);
    }
    #pragma unroll
    for (int i = 0; i < 4; ++i) {
        const int u   = tid + i * 128;
        const int d   = u >> 3;
        const int seg = u & 7;
        *(uint4*)(Vbuf + d * 64 + seg * 8) =
            *(const uint4*)(Vh + (long)d * 4096 + kb + seg * 8);
    }
#endif
}

// ---------------------------------------------------------------------------
// Flash attention (transposed, base-2 softmax), 64-key tiles, double-buffered
// async-to-LDS pipeline. One block = (b, head, 64-query tile), 4 waves.
// Q,K: [head][s][d] bf16 ; V: [head][d][s] bf16.
// ---------------------------------------------------------------------------
__global__ __launch_bounds__(128)
void attn_kernel(const unsigned short* __restrict__ Q,
                 const unsigned short* __restrict__ K,
                 const unsigned short* __restrict__ V,
                 float* __restrict__ out)                 // [4][1048576] f32
{
    __shared__ unsigned short Klds[2][64 * 64];   // [buf][k][d] bf16 (16 KB)
    __shared__ unsigned short Vtl[2][64 * 64];    // [buf][d][k] bf16 (16 KB)

    const int tid  = threadIdx.x;
    const int lane = tid & 31;
    const int wv   = tid >> 5;
    const int hi   = (lane >> 4) & 1;
    const int ln   = lane & 15;

    const int qt = blockIdx.x;           // query tile (64 queries)
    const int n  = blockIdx.y;           // head
    const int b  = blockIdx.z;           // batch
    const long head = (long)(b * 4 + n);

    const unsigned short* Qh = Q + head * (4096L * 64);
    const unsigned short* Kh = K + head * (4096L * 64);
    const unsigned short* Vh = V + head * (4096L * 64);  // [d][s]

    const int qg = qt * 64 + wv * 16 + ln;   // this lane's query row

    // Loop-invariant B operands: Q^T halves (Kdim = d).  B: k = hi*16 + e.
    const unsigned short* qrow = Qh + (long)qg * 64;
    BF16x16 Qt0, Qt1;
    Qt0.q[0] = *(const uint4*)(qrow + hi * 16);
    Qt0.q[1] = *(const uint4*)(qrow + hi * 16 + 8);
    Qt1.q[0] = *(const uint4*)(qrow + 32 + hi * 16);
    Qt1.q[1] = *(const uint4*)(qrow + 32 + hi * 16 + 8);

    v8f accO[4] = {v8f{}, v8f{}, v8f{}, v8f{}};
    float m_run = -3.0e38f;
    float l_run = 0.0f;

    // Prologue: stage tile 0 into buffer 0
    stage_tiles(Kh, Vh, 0, tid, Klds[0], Vtl[0]);

    for (int it = 0; it < 64; ++it) {
        const int kb  = it * 64;
        const int cur = it & 1;

#if USE_ASYNC_LDS
        async_wait0();       // tile `it` has landed in LDS
#endif
        __syncthreads();     // visible to all waves; prior reads of alt buf done

        // Pipeline: kick off DMA of tile it+1 into the other buffer
        if (kb + 64 < 4096)
            stage_tiles(Kh, Vh, kb + 64, tid, Klds[cur ^ 1], Vtl[cur ^ 1]);
        // Warm L2 two tiles ahead (one 128B line per thread)
        if (kb + 128 < 4096) {
            if (tid < 64)
                __builtin_prefetch(Kh + (long)(kb + 128) * 64 + tid * 64, 0, 3);
            else
                __builtin_prefetch(Vh + (long)(tid - 64) * 4096 + kb + 128, 0, 3);
        }

        const unsigned short* Kc = Klds[cur];
        const unsigned short* Vc = Vtl[cur];

        // --- S^T: 4 key tiles of 16 (M=key, N=query), Kdim=d (2x32) ---
        v8f st[4];
        #pragma unroll
        for (int t = 0; t < 4; ++t) {
            v8f sv = {};
            BF16x16 A;
            A.q[0] = *(const uint4*)&Kc[(t * 16 + ln) * 64 + hi * 16];
            A.q[1] = *(const uint4*)&Kc[(t * 16 + ln) * 64 + 8 + hi * 16];
            sv = __builtin_amdgcn_wmma_f32_16x16x32_bf16(false, A.v, false, Qt0.v, (short)0, sv, false, false);
            A.q[0] = *(const uint4*)&Kc[(t * 16 + ln) * 64 + 32 + hi * 16];
            A.q[1] = *(const uint4*)&Kc[(t * 16 + ln) * 64 + 40 + hi * 16];
            sv = __builtin_amdgcn_wmma_f32_16x16x32_bf16(false, A.v, false, Qt1.v, (short)0, sv, false, false);
            st[t] = sv;
        }

        // --- Online softmax in 2^x domain (per query col, combine halves) ---
        float mx = m_run;
        #pragma unroll
        for (int t = 0; t < 4; ++t)
            #pragma unroll
            for (int r = 0; r < 8; ++r) mx = fmaxf(mx, st[t][r]);
        mx = fmaxf(mx, __shfl_xor(mx, 16, 32));
        const float scale = exp2f(m_run - mx);
        m_run = mx;

        float rs = 0.0f;
        #pragma unroll
        for (int t = 0; t < 4; ++t)
            #pragma unroll
            for (int r = 0; r < 8; ++r) {
                const float pv = exp2f(st[t][r] - mx);
                st[t][r] = pv;
                rs += pv;
            }
        rs += __shfl_xor(rs, 16, 32);
        l_run = l_run * scale + rs;

        #pragma unroll
        for (int dt = 0; dt < 4; ++dt)
            #pragma unroll
            for (int r = 0; r < 8; ++r) accO[dt][r] *= scale;

        // --- Re-layout P as two B operands (Kdim=key 32 each, N=query) ---
        BF16x16 Pb0, Pb1;
        #pragma unroll
        for (int r = 0; r < 8; ++r) {
            const float a0 = __shfl_xor(st[0][r], 16, 32);
            const float a1 = __shfl_xor(st[1][r], 16, 32);
            Pb0.s[r]     = f2bf(hi ? a1 : st[0][r]);
            Pb0.s[8 + r] = f2bf(hi ? st[1][r] : a0);
            const float a2 = __shfl_xor(st[2][r], 16, 32);
            const float a3 = __shfl_xor(st[3][r], 16, 32);
            Pb1.s[r]     = f2bf(hi ? a3 : st[2][r]);
            Pb1.s[8 + r] = f2bf(hi ? st[3][r] : a2);
        }

        // --- O^T += V^T * P  (A: M=d tile of 16, Kdim=key 2x32) ---
        #pragma unroll
        for (int kc = 0; kc < 2; ++kc) {
            #pragma unroll
            for (int dt = 0; dt < 4; ++dt) {
                BF16x16 Av;
                const int d = dt * 16 + ln;
                Av.q[0] = *(const uint4*)&Vc[d * 64 + kc * 32 + hi * 16];
                Av.q[1] = *(const uint4*)&Vc[d * 64 + kc * 32 + 8 + hi * 16];
                accO[dt] = __builtin_amdgcn_wmma_f32_16x16x32_bf16(
                    false, Av.v, false, kc ? Pb1.v : Pb0.v,
                    (short)0, accO[dt], false, false);
            }
        }
    }

    // --- Epilogue: /l, store. O^T C-layout: lane=q, VGPR r -> d=dt*16+r+8*hi
    const float inv = 1.0f / l_run;
    float* ob = out + (long)b * 1048576 + (long)qg * 256 + n * 64;
    #pragma unroll
    for (int dt = 0; dt < 4; ++dt) {
        float4 lo = make_float4(accO[dt][0] * inv, accO[dt][1] * inv,
                                accO[dt][2] * inv, accO[dt][3] * inv);
        float4 hi4 = make_float4(accO[dt][4] * inv, accO[dt][5] * inv,
                                 accO[dt][6] * inv, accO[dt][7] * inv);
        *(float4*)(ob + dt * 16 + hi * 8)     = lo;
        *(float4*)(ob + dt * 16 + hi * 8 + 4) = hi4;
    }
}

// ---------------------------------------------------------------------------
extern "C" void kernel_launch(void* const* d_in, const int* in_sizes, int n_in,
                              void* d_out, int out_size, void* d_ws, size_t ws_size,
                              hipStream_t stream) {
    (void)in_sizes; (void)n_in; (void)out_size; (void)ws_size;

    const float* x     = (const float*)d_in[0];
    const float* wq    = (const float*)d_in[1];
    const float* bq    = (const float*)d_in[2];
    const float* wk    = (const float*)d_in[3];
    const float* bk    = (const float*)d_in[4];
    const float* wv    = (const float*)d_in[5];
    const float* bv    = (const float*)d_in[6];
    const float* rel_h = (const float*)d_in[7];
    const float* rel_w = (const float*)d_in[8];
    float* out = (float*)d_out;

    // Workspace layout (bf16 elements):
    //   [0]        wqb/wkb/wvb : 3 * 65536
    //   [196608]   xT          : 4,194,304
    //   [+4M]      Q, K', V^T  : 4,194,304 each   (~34 MB total)
    unsigned short* wsp = (unsigned short*)d_ws;
    unsigned short* wqb = wsp;
    unsigned short* wkb = wqb + 65536ull;
    unsigned short* wvb = wkb + 65536ull;
    unsigned short* xT  = wvb + 65536ull;
    unsigned short* qws = xT  + 4194304ull;
    unsigned short* kws = qws + 4194304ull;
    unsigned short* vws = kws + 4194304ull;

    cvt_kernel<<<256, 256, 0, stream>>>(wq, wqb, 65536);
    cvt_kernel<<<256, 256, 0, stream>>>(wk, wkb, 65536);
    cvt_kernel<<<256, 256, 0, stream>>>(wv, wvb, 65536);
    xpose_kernel<<<16384, 256, 0, stream>>>(x, xT);

    const dim3 pgrid(16, 32);
    for (int b = 0; b < 4; ++b) {
        const unsigned short* xTb = xT + (long)b * 1048576;
        unsigned short* qb = qws + (long)b * 1048576;
        unsigned short* kb = kws + (long)b * 1048576;
        unsigned short* vb = vws + (long)b * 1048576;
        proj_kernel<<<pgrid, 256, 0, stream>>>(xTb, wqb, bq, rel_h, rel_w, 0, qb);
        proj_kernel<<<pgrid, 256, 0, stream>>>(xTb, wkb, bk, rel_h, rel_w, 1, kb);
        proj_kernel<<<pgrid, 256, 0, stream>>>(xTb, wvb, bv, rel_h, rel_w, 2, vb);
    }

    attn_kernel<<<dim3(64, 4, 4), 128, 0, stream>>>(qws, kws, vws, out);
}
